// Network_8143257993998
// MI455X (gfx1250) — matrix operand order repaired and checked by reading, not compiled, not run on hardware
//
#include <hip/hip_runtime.h>
#include <hip/hip_bf16.h>

// ---------------------------------------------------------------- types
typedef __bf16 v16bf __attribute__((ext_vector_type(16)));
typedef float  v8f   __attribute__((ext_vector_type(8)));
typedef unsigned int u32x4 __attribute__((ext_vector_type(4)));
typedef int i32x4 __attribute__((ext_vector_type(4)));
typedef int i32x8 __attribute__((ext_vector_type(8)));
union AFrag { unsigned u[8]; v16bf v; };

__device__ __forceinline__ unsigned pk_bf16(float a, float b) {
  unsigned short ua = __builtin_bit_cast(unsigned short, (__bf16)a);
  unsigned short ub = __builtin_bit_cast(unsigned short, (__bf16)b);
  return (unsigned)ua | ((unsigned)ub << 16);
}
__device__ __forceinline__ float gelu_f(float x) {
  return 0.5f * x * (1.0f + erff(x * 0.70710678118654752f));
}

// ---------------------------------------------------------------- generic aligned WMMA GEMM
// C[M,N] = act(A[M,K] @ B^T + bias) (+ addsrc).  B row-major (N,K).
// REQUIRES: M,N,K multiples of 32; A/B rows 16B-aligned.
__global__ void k_gemm(const float* __restrict__ A, int lda,
                       const float* __restrict__ B, int ldb,
                       const float* __restrict__ bias,
                       const float* __restrict__ addsrc, int ldadd,
                       float* __restrict__ C, int ldc,
                       int M, int N, int K, int act)
{
  // packed bf16 pairs along K: [row 0..31][k-dword 0..15] (+2 pad dwords)
  __shared__ unsigned Ap[32][18];
  __shared__ unsigned Bp[32][18];
  const int tid  = threadIdx.x;      // 128 threads = 4 waves
  const int wave = tid >> 5;
  const int lane = tid & 31;
  const int m0 = blockIdx.y * 32;
  const int n0 = blockIdx.x * 32;
  const int wm = (wave & 1) * 16;
  const int wn = (wave >> 1) * 16;

  v8f acc = {0.f, 0.f, 0.f, 0.f, 0.f, 0.f, 0.f, 0.f};

  for (int k0 = 0; k0 < K; k0 += 32) {
#pragma unroll
    for (int i = 0; i < 2; ++i) {
      int v = tid + i * 128;          // 256 float4 transactions per matrix
      int r = v >> 3, f4 = v & 7;
      float4 a = *(const float4*)&A[(size_t)(m0 + r) * lda + k0 + f4 * 4];
      Ap[r][f4 * 2]     = pk_bf16(a.x, a.y);
      Ap[r][f4 * 2 + 1] = pk_bf16(a.z, a.w);
      float4 b = *(const float4*)&B[(size_t)(n0 + r) * ldb + k0 + f4 * 4];
      Bp[r][f4 * 2]     = pk_bf16(b.x, b.y);
      Bp[r][f4 * 2 + 1] = pk_bf16(b.z, b.w);
    }
    if (k0 + 32 < K)                  // global_prefetch_b8 on the K-stream
      __builtin_prefetch(&A[(size_t)(m0 + (tid >> 2)) * lda + k0 + 32], 0, 1);
    __syncthreads();

    AFrag af, bf;
    const int ar = wm + (lane & 15);
    const int ha = (lane >> 4) * 4;   // A dword-halves per CDNA5 layout
#pragma unroll
    for (int j = 0; j < 4; ++j) {
      af.u[j]     = Ap[ar][ha + j];       // K = {0..7} / {8..15}
      af.u[j + 4] = Ap[ar][8 + ha + j];   // K = {16..23} / {24..31}
    }
    const int bc = wn + (lane & 15);
    const int hb = (lane >> 4) * 4;
#pragma unroll
    for (int j = 0; j < 8; ++j) bf.u[j] = Bp[bc][hb + (j & 3) + (j >> 2) * 8];

    acc = __builtin_amdgcn_wmma_f32_16x16x32_bf16(false, af.v, false, bf.v,
                                                  (short)0, acc, false, false);
    __syncthreads();
  }

  const int colg = n0 + wn + (lane & 15);
#pragma unroll
  for (int j = 0; j < 8; ++j) {
    int rowg = m0 + wm + j + 8 * (lane >> 4);
    float v = acc[j];
    if (bias)   v += bias[colg];
    if (act)    v = gelu_f(v);
    if (addsrc) v += addsrc[(size_t)rowg * ldadd + colg];
    C[(size_t)rowg * ldc + colg] = v;
  }
}

// ---------------------------------------------------------------- TDM-staged WMMA GEMM (fc3)
// Double-buffered: Tensor Data Mover DMA of tile i+1 overlaps WMMA on tile i.
__device__ __forceinline__ void tdm_load_tile_f32(unsigned lds_addr,
                                                  const float* gsrc,
                                                  unsigned dim0, unsigned dim1,
                                                  unsigned stride0)
{
  unsigned long long ga = (unsigned long long)(size_t)gsrc;
  u32x4 g0; i32x8 g1; i32x4 g2, g3;
  g0[0] = 1u;                                    // count=1, user mode
  g0[1] = lds_addr;                              // LDS byte address
  g0[2] = (unsigned)ga;                          // global_addr[31:0]
  g0[3] = (unsigned)(ga >> 32) | (2u << 30);     // global_addr[56:32] | type=2
  g1[0] = (int)(2u << 16);                       // data_size=4B, no multicast/pad
  g1[1] = (int)((dim0 & 0xFFFFu) << 16);         // tensor_dim0[15:0]
  g1[2] = (int)(((dim0 >> 16) & 0xFFFFu) | ((dim1 & 0xFFFFu) << 16));
  g1[3] = (int)(((dim1 >> 16) & 0xFFFFu) | (32u << 16));   // tile_dim0=32
  g1[4] = 32;                                    // tile_dim1=32, tile_dim2=0
  g1[5] = (int)stride0;                          // tensor_dim0_stride[31:0]
  g1[6] = 0;                                     // stride0[47:32], dim1_stride lo
  g1[7] = 0;
  g2[0] = 0; g2[1] = 0; g2[2] = 0; g2[3] = 0;    // 2-D: groups 2/3 unused
  g3[0] = 0; g3[1] = 0; g3[2] = 0; g3[3] = 0;
#if defined(__clang_major__) && (__clang_major__ >= 23)
  i32x8 z8 = {0, 0, 0, 0, 0, 0, 0, 0};
  __builtin_amdgcn_tensor_load_to_lds(g0, g1, g2, g3, z8, 0);
#else
  __builtin_amdgcn_tensor_load_to_lds(g0, g1, g2, g3, 0);
#endif
}

__global__ void k_gemm_tdm(const float* __restrict__ A, int lda,
                           const float* __restrict__ B, int ldb,
                           const float* __restrict__ bias,
                           float* __restrict__ C, int ldc,
                           int M, int N, int K, int act)
{
  __shared__ float As[2][32][32];
  __shared__ float Bs[2][32][32];
  const int tid  = threadIdx.x;      // 128 threads = 4 waves
  const int wave = tid >> 5;
  const int lane = tid & 31;
  const int m0 = blockIdx.y * 32;
  const int n0 = blockIdx.x * 32;
  const int wm = (wave & 1) * 16;
  const int wn = (wave >> 1) * 16;
  const int nit = K / 32;

  unsigned ldsA[2], ldsB[2];
  ldsA[0] = (unsigned)(size_t)&As[0][0][0];
  ldsA[1] = (unsigned)(size_t)&As[1][0][0];
  ldsB[0] = (unsigned)(size_t)&Bs[0][0][0];
  ldsB[1] = (unsigned)(size_t)&Bs[1][0][0];

  v8f acc = {0.f, 0.f, 0.f, 0.f, 0.f, 0.f, 0.f, 0.f};

  if (tid < 32) {                     // wave-uniform: preload tile 0
    tdm_load_tile_f32(ldsA[0], &A[(size_t)m0 * lda], (unsigned)K, (unsigned)M, (unsigned)lda);
    tdm_load_tile_f32(ldsB[0], &B[(size_t)n0 * ldb], (unsigned)K, (unsigned)N, (unsigned)ldb);
  }

  for (int it = 0; it < nit; ++it) {
    const int buf = it & 1;
    if (tid < 32) {
      if (it + 1 < nit) {             // issue next tile, then wait for current
        int k1 = (it + 1) * 32;
        tdm_load_tile_f32(ldsA[buf ^ 1], &A[(size_t)m0 * lda + k1],
                          (unsigned)K, (unsigned)M, (unsigned)lda);
        tdm_load_tile_f32(ldsB[buf ^ 1], &B[(size_t)n0 * ldb + k1],
                          (unsigned)K, (unsigned)N, (unsigned)ldb);
        __builtin_amdgcn_s_wait_tensorcnt(2);   // in-order: current tiles done
      } else {
        __builtin_amdgcn_s_wait_tensorcnt(0);
      }
    }
    __syncthreads();                  // tile `buf` visible to all waves

    AFrag af, bf;
    const int ar = wm + (lane & 15);
    const int kb = (lane >> 4) * 8;
#pragma unroll
    for (int j = 0; j < 8; ++j) {
      int kk = (j < 4) ? (kb + 2 * j) : (16 + kb + 2 * (j - 4));
      af.u[j] = pk_bf16(As[buf][ar][kk], As[buf][ar][kk + 1]);
    }
    const int bc = wn + (lane & 15);
    const int kbb = (lane >> 4) * 8;
#pragma unroll
    for (int j = 0; j < 8; ++j) {
      int kk = (j < 4) ? (kbb + 2 * j) : (16 + kbb + 2 * (j - 4));
      bf.u[j] = pk_bf16(Bs[buf][bc][kk], Bs[buf][bc][kk + 1]);
    }
    acc = __builtin_amdgcn_wmma_f32_16x16x32_bf16(false, af.v, false, bf.v,
                                                  (short)0, acc, false, false);
    __syncthreads();                  // done with `buf` before TDM reuses it
  }

  const int colg = n0 + wn + (lane & 15);
#pragma unroll
  for (int j = 0; j < 8; ++j) {
    int rowg = m0 + wm + j + 8 * (lane >> 4);
    float v = acc[j] + bias[colg];
    if (act) v = gelu_f(v);
    C[(size_t)rowg * ldc + colg] = v;
  }
}

// ---------------------------------------------------------------- implicit-GEMM dilated conv (64->64 ch, k=3) + GELU
// X,Y layout: [(n*64 + ch)*512 + l].  W: (64 out, 64 in, 3).  K = 192 (t-major, 6 steps of 32).
__global__ void k_conv_wmma(const float* __restrict__ X,
                            const float* __restrict__ W,
                            const float* __restrict__ bias,
                            float* __restrict__ Y, int dil)
{
  __shared__ unsigned Ap[32][18];     // [l][cin-pair]
  __shared__ unsigned Bp[32][18];     // [cout][cin-pair]
  const int tid  = threadIdx.x;       // 128
  const int wave = tid >> 5;
  const int lane = tid & 31;
  const int n   = blockIdx.y;
  const int ncb = blockIdx.x & 1;
  const int lb  = blockIdx.x >> 1;
  const int nc0 = ncb * 32;
  const int l0  = lb * 32;
  const int wm = (wave & 1) * 16;
  const int wn = (wave >> 1) * 16;

  v8f acc = {0.f, 0.f, 0.f, 0.f, 0.f, 0.f, 0.f, 0.f};

  for (int ks = 0; ks < 6; ++ks) {
    const int t    = ks >> 1;
    const int cinb = (ks & 1) * 32;
#pragma unroll
    for (int i = 0; i < 4; ++i) {
      int v = tid + i * 128;          // 512 pair-slots
      int li = v & 31, kp = v >> 5;   // kp in [0,16)
      int pos  = l0 + li + (t - 1) * dil;
      int cin0 = cinb + kp * 2;
      float x0 = 0.f, x1 = 0.f;
      if (pos >= 0 && pos < 512) {
        const float* xp = &X[((size_t)n * 64 + cin0) * 512 + pos];
        x0 = xp[0];
        x1 = xp[512];
      }
      Ap[li][kp] = pk_bf16(x0, x1);
      const float* wp = &W[(size_t)(nc0 + li) * 192 + (cinb + kp * 2) * 3 + t];
      Bp[li][kp] = pk_bf16(wp[0], wp[3]);
    }
    __syncthreads();

    AFrag af, bf;
    const int ar = wm + (lane & 15);
    const int ha = (lane >> 4) * 4;
#pragma unroll
    for (int j = 0; j < 4; ++j) {
      af.u[j]     = Ap[ar][ha + j];
      af.u[j + 4] = Ap[ar][8 + ha + j];
    }
    const int bc = wn + (lane & 15);
    const int hb = (lane >> 4) * 4;
#pragma unroll
    for (int j = 0; j < 8; ++j) bf.u[j] = Bp[bc][hb + (j & 3) + (j >> 2) * 8];

    acc = __builtin_amdgcn_wmma_f32_16x16x32_bf16(false, af.v, false, bf.v,
                                                  (short)0, acc, false, false);
    __syncthreads();
  }

  const int col = wn + (lane & 15);
#pragma unroll
  for (int j = 0; j < 8; ++j) {
    int row = wm + j + 8 * (lane >> 4);
    float v = gelu_f(acc[j] + bias[nc0 + col]);
    Y[((size_t)n * 64 + nc0 + col) * 512 + l0 + row] = v;
  }
}

// ---------------------------------------------------------------- small kernels
// patches, zero-padded K: Sp[(n*64+p)*32+k] = k<16 ? s_pad[n, p*8+k] : 0
__global__ void k_patches(const float* __restrict__ s, float* __restrict__ Sp)
{
  int i = blockIdx.x * 256 + threadIdx.x;          // 32768*32
  if (i >= 32768 * 32) return;
  int k = i & 31, row = i >> 5;
  int p = row & 63, n = row >> 6;
  int b = n >> 5, c = n & 31;
  float v = 0.f;
  if (k < 16) {
    int pos = p * 8 + k; if (pos > 511) pos = 511;  // edge pad
    v = s[((size_t)b * 512 + pos) * 32 + c];
  }
  Sp[i] = v;
}

// zero-pad fc1_w (256,16) -> (256,32)
__global__ void k_padw(const float* __restrict__ w, float* __restrict__ wp)
{
  int i = blockIdx.x * 256 + threadIdx.x;          // 256*32
  if (i >= 256 * 32) return;
  int j = i & 31, o = i >> 5;
  wp[i] = (j < 16) ? w[o * 16 + j] : 0.f;
}

// t2[n*512+i] = t[b,i,c]
__global__ void k_t2(const float* __restrict__ t, float* __restrict__ t2)
{
  int i = blockIdx.x * 256 + threadIdx.x;          // 512*512
  if (i >= 512 * 512) return;
  int l = i & 511, n = i >> 9;
  int b = n >> 5, c = n & 31;
  t2[i] = t[((size_t)b * 512 + l) * 32 + c];
}

// LayerNorm over 256 features, one row per block (256 threads)
__global__ void k_layernorm(const float* __restrict__ X,
                            const float* __restrict__ g, const float* __restrict__ b,
                            float* __restrict__ Y)
{
  __shared__ float red[256];
  int row = blockIdx.x, t = threadIdx.x;
  float x = X[(size_t)row * 256 + t];
  red[t] = x; __syncthreads();
  for (int s = 128; s > 0; s >>= 1) { if (t < s) red[t] += red[t + s]; __syncthreads(); }
  float m = red[0] * (1.0f / 256.0f); __syncthreads();
  float dx = x - m;
  red[t] = dx * dx; __syncthreads();
  for (int s = 128; s > 0; s >>= 1) { if (t < s) red[t] += red[t + s]; __syncthreads(); }
  float v = red[0] * (1.0f / 256.0f);
  Y[(size_t)row * 256 + t] = dx * rsqrtf(v + 1e-5f) * g[t] + b[t];
}

// depthwise conv k=3 pad=1 over p (64), channels 256, + GELU
__global__ void k_dwconv(const float* __restrict__ X,
                         const float* __restrict__ w, const float* __restrict__ b,
                         float* __restrict__ Y)
{
  int i = blockIdx.x * 256 + threadIdx.x;          // 32768*256
  if (i >= 32768 * 256) return;
  int d = i & 255, np = i >> 8;
  int p = np & 63, n = np >> 6;
  float acc = b[d];
#pragma unroll
  for (int t = 0; t < 3; ++t) {
    int pp = p + t - 1;
    if (pp >= 0 && pp < 64)
      acc += w[d * 3 + t] * X[(((size_t)n * 64 + pp) * 256) + d];
  }
  Y[i] = gelu_f(acc);
}

// first encoder conv (1 -> 64 ch) + GELU
__global__ void k_conv1(const float* __restrict__ t2,
                        const float* __restrict__ w, const float* __restrict__ b,
                        float* __restrict__ Y, int dil)
{
  int i = blockIdx.x * 256 + threadIdx.x;          // 512*64*512
  if (i >= 512 * 64 * 512) return;
  int l = i & 511, co = (i >> 9) & 63, n = i >> 15;
  float acc = b[co];
#pragma unroll
  for (int t = 0; t < 3; ++t) {
    int pos = l + (t - 1) * dil;
    if (pos >= 0 && pos < 512) acc += w[co * 3 + t] * t2[(size_t)n * 512 + pos];
  }
  Y[i] = gelu_f(acc);
}

// batch-norm statistics over (n,l) per channel. 64 blocks x 256 threads.
__global__ void k_bnstats(const float* __restrict__ Y, float* __restrict__ bm,
                          float* __restrict__ bv)
{
  __shared__ float r1[256], r2[256];
  int ch = blockIdx.x, t = threadIdx.x;
  float s = 0.f, ss = 0.f;
  for (int i = t; i < 512 * 512; i += 256) {
    int n = i >> 9, l = i & 511;
    float x = Y[((size_t)n * 64 + ch) * 512 + l];
    s += x; ss += x * x;
  }
  r1[t] = s; r2[t] = ss; __syncthreads();
  for (int k = 128; k > 0; k >>= 1) {
    if (t < k) { r1[t] += r1[t + k]; r2[t] += r2[t + k]; }
    __syncthreads();
  }
  if (t == 0) {
    float m = r1[0] * (1.0f / 262144.0f);
    bm[ch] = m;
    bv[ch] = r2[0] * (1.0f / 262144.0f) - m * m;
  }
}

__global__ void k_bnapply(float* __restrict__ Y, const float* __restrict__ bm,
                          const float* __restrict__ bv,
                          const float* __restrict__ g, const float* __restrict__ be)
{
  int i = blockIdx.x * 256 + threadIdx.x;          // 512*64*512
  if (i >= 512 * 64 * 512) return;
  int ch = (i >> 9) & 63;
  Y[i] = (Y[i] - bm[ch]) * rsqrtf(bv[ch] + 1e-5f) * g[ch] + be[ch];
}

// attention pooling: logits(64->4), softmax over L=512, nodes = attn^T @ feats
__global__ void k_pool(const float* __restrict__ X,        // (n,64,512)
                       const float* __restrict__ pw, const float* __restrict__ pb,
                       float* __restrict__ nodes, int soff)
{
  __shared__ float lg[512][4];
  __shared__ float red[256];
  __shared__ float sm4[4];
  int n = blockIdx.x, t = threadIdx.x;

  for (int li = t; li < 512; li += 256) {
    float a0 = pb[0], a1 = pb[1], a2 = pb[2], a3 = pb[3];
    for (int ch = 0; ch < 64; ++ch) {
      float xv = X[((size_t)n * 64 + ch) * 512 + li];
      a0 += xv * pw[0 * 64 + ch];
      a1 += xv * pw[1 * 64 + ch];
      a2 += xv * pw[2 * 64 + ch];
      a3 += xv * pw[3 * 64 + ch];
    }
    lg[li][0] = a0; lg[li][1] = a1; lg[li][2] = a2; lg[li][3] = a3;
  }
  __syncthreads();

  for (int k = 0; k < 4; ++k) {
    red[t] = fmaxf(lg[t][k], lg[t + 256][k]); __syncthreads();
    for (int s = 128; s > 0; s >>= 1) { if (t < s) red[t] = fmaxf(red[t], red[t + s]); __syncthreads(); }
    float mx = red[0]; __syncthreads();
    float e0 = expf(lg[t][k] - mx), e1 = expf(lg[t + 256][k] - mx);
    lg[t][k] = e0; lg[t + 256][k] = e1;
    red[t] = e0 + e1; __syncthreads();
    for (int s = 128; s > 0; s >>= 1) { if (t < s) red[t] += red[t + s]; __syncthreads(); }
    if (t == 0) sm4[k] = red[0];
    __syncthreads();
  }

  int k = t >> 6, d = t & 63;
  float acc = 0.f;
  for (int li = 0; li < 512; ++li)
    acc += lg[li][k] * X[((size_t)n * 64 + d) * 512 + li];
  nodes[((size_t)n * 12 + soff + k) * 64 + d] = acc / sm4[k];
}

// DTW distances + softmax memory read + node mean -> fused (512 x 128)
__global__ void k_dtw_fuse(const float* __restrict__ nodes,
                           const float* __restrict__ protos,
                           const float* __restrict__ mem_vals,
                           float* __restrict__ fused)
{
  __shared__ float nd[12][64];
  __shared__ float qn[12];
  __shared__ float dst[64];
  const float INF = 1e9f;
  int n = blockIdx.x, t = threadIdx.x;   // 64 threads

  for (int q = 0; q < 12; ++q) nd[q][t] = nodes[((size_t)n * 12 + q) * 64 + t];
  __syncthreads();
  if (t < 12) {
    float s = 0.f;
    for (int d = 0; d < 64; ++d) s += nd[t][d] * nd[t][d];
    qn[t] = s;
  }
  __syncthreads();

  const int m = t;
  float pn[8];
#pragma unroll
  for (int p = 0; p < 8; ++p) {
    const float* pr = &protos[((size_t)m * 8 + p) * 64];
    float s = 0.f;
    for (int d = 0; d < 64; ++d) s += pr[d] * pr[d];
    pn[p] = s;
  }
  float prev[9], cur[9];
  prev[0] = 0.f;
#pragma unroll
  for (int p = 1; p < 9; ++p) prev[p] = INF;
  for (int q = 0; q < 12; ++q) {
    cur[0] = INF;
    for (int p = 1; p < 9; ++p) {
      const float* pr = &protos[((size_t)m * 8 + (p - 1)) * 64];
      float ip = 0.f;
      for (int d = 0; d < 64; ++d) ip += nd[q][d] * pr[d];
      float cst = sqrtf(fmaxf(qn[q] + pn[p - 1] - 2.0f * ip, 1e-12f));
      cur[p] = cst + fminf(fminf(prev[p], prev[p - 1]), cur[p - 1]);
    }
#pragma unroll
    for (int p = 0; p < 9; ++p) prev[p] = cur[p];
  }
  dst[m] = prev[8];
  __syncthreads();

  float mx = -1e30f;
  for (int mm = 0; mm < 64; ++mm) mx = fmaxf(mx, -dst[mm]);
  float sm = 0.f;
  for (int mm = 0; mm < 64; ++mm) sm += expf(-dst[mm] - mx);
  float rd = 0.f;
  for (int mm = 0; mm < 64; ++mm)
    rd += (expf(-dst[mm] - mx) / sm) * mem_vals[(size_t)mm * 64 + t];
  float mean = 0.f;
  for (int q = 0; q < 12; ++q) mean += nd[q][t];
  mean *= (1.0f / 12.0f);
  fused[(size_t)n * 128 + t]      = mean;
  fused[(size_t)n * 128 + 64 + t] = rd;
}

// final transpose: out[b, j, c] = x[b*32+c, j]
__global__ void k_final(const float* __restrict__ xm, float* __restrict__ out)
{
  int i = blockIdx.x * 256 + threadIdx.x;          // 16*96*32
  if (i >= 16 * 96 * 32) return;
  int c = i & 31, j = (i / 32) % 96, b = i / (96 * 32);
  out[i] = xm[((size_t)b * 32 + c) * 96 + j];
}

// ---------------------------------------------------------------- host launcher
static inline void gemm(hipStream_t st, const float* A, int lda, const float* B, int ldb,
                        const float* bias, const float* addsrc, int ldadd,
                        float* C, int ldc, int M, int N, int K, int act)
{
  dim3 g(N / 32, M / 32), b(128);
  k_gemm<<<g, b, 0, st>>>(A, lda, B, ldb, bias, addsrc, ldadd, C, ldc, M, N, K, act);
}

extern "C" void kernel_launch(void* const* d_in, const int* in_sizes, int n_in,
                              void* d_out, int out_size, void* d_ws, size_t ws_size,
                              hipStream_t stream)
{
  (void)in_sizes; (void)n_in; (void)out_size; (void)ws_size;
  const float* s     = (const float*)d_in[0];
  const float* t     = (const float*)d_in[1];
  const float* fc1_w = (const float*)d_in[2];
  const float* fc1_b = (const float*)d_in[3];
  const float* ln1_g = (const float*)d_in[4];
  const float* ln1_b = (const float*)d_in[5];
  const float* dw_w  = (const float*)d_in[6];
  const float* dw_b  = (const float*)d_in[7];
  const float* pw_w  = (const float*)d_in[8];
  const float* pw_b  = (const float*)d_in[9];
  const float* ln2_g = (const float*)d_in[10];
  const float* ln2_b = (const float*)d_in[11];
  const float* res_w = (const float*)d_in[12];
  const float* res_b = (const float*)d_in[13];
  const float* fc3_w = (const float*)d_in[14];
  const float* fc3_b = (const float*)d_in[15];
  const float* fc4_w = (const float*)d_in[16];
  const float* fc4_b = (const float*)d_in[17];
  const float* fc8_w = (const float*)d_in[18];
  const float* fc8_b = (const float*)d_in[19];
  const float* protos   = (const float*)d_in[74];
  const float* mem_vals = (const float*)d_in[75];
  const float* rp1_w    = (const float*)d_in[76];
  const float* rp1_b    = (const float*)d_in[77];
  const float* rp2_w    = (const float*)d_in[78];
  const float* rp2_b    = (const float*)d_in[79];

  // -------- workspace layout --------
  const size_t MB = 1024 * 1024;
  char* base = (char*)d_ws;
  float* Xcat    = (float*)base;                 // 512*192
  float* G1      = Xcat + 512 * 192;             // 512*192
  float* fused   = G1 + 512 * 192;               // 512*128
  float* t_mid   = fused + 512 * 128;            // 512*64
  float* out_mat = t_mid + 512 * 64;             // 512*96
  float* bn_m    = out_mat + 512 * 96;           // 64
  float* bn_v    = bn_m + 64;                    // 64
  float* nodes   = bn_v + 64;                    // 512*12*64
  float* t2      = nodes + 512 * 12 * 64;        // 512*512
  float* fc1wp   = t2 + 512 * 512;               // 256*32  (all < 8 MB)
  // big region @ 8 MB; s-branch and t-branch overlap (stream-serialized)
  float* Sp   = (float*)(base + 8 * MB);         // 32768*32  (4 MB)
  float* bufA = (float*)(base + 12 * MB);        // 32 MB
  float* s_e  = (float*)(base + 44 * MB);        // 32 MB
  float* bufB = (float*)(base + 76 * MB);        // 32 MB
  float* E0   = (float*)(base + 8 * MB);         // 64 MB (t-branch ping)
  float* E1   = (float*)(base + 72 * MB);        // 64 MB (t-branch pong)

  // ================= s-branch =================
  k_patches<<<dim3((32768 * 32) / 256), dim3(256), 0, stream>>>(s, Sp);
  k_padw<<<dim3(32), dim3(256), 0, stream>>>(fc1_w, fc1wp);
  // fc1 + gelu : (32768,32) @ (256,32)^T -> bufA
  gemm(stream, Sp, 32, fc1wp, 32, fc1_b, nullptr, 0, bufA, 256, 32768, 256, 32, 1);
  k_layernorm<<<dim3(32768), dim3(256), 0, stream>>>(bufA, ln1_g, ln1_b, s_e);
  k_dwconv<<<dim3(32768), dim3(256), 0, stream>>>(s_e, dw_w, dw_b, bufB);
  // pointwise conv + gelu
  gemm(stream, bufB, 256, pw_w, 256, pw_b, nullptr, 0, bufA, 256, 32768, 256, 256, 1);
  k_layernorm<<<dim3(32768), dim3(256), 0, stream>>>(bufA, ln2_g, ln2_b, bufB);
  // residual: LN2(h) + linear(s_e)
  gemm(stream, s_e, 256, res_w, 256, res_b, bufB, 256, bufA, 256, 32768, 256, 256, 0);
  // fc3 + gelu via double-buffered Tensor Data Mover staging
  k_gemm_tdm<<<dim3(192 / 32, 512 / 32), dim3(128), 0, stream>>>(
      bufA, 16384, fc3_w, 16384, fc3_b, G1, 192, 512, 192, 16384, 1);
  // fc4 -> s_head into Xcat[:, 0:96]
  gemm(stream, G1, 192, fc4_w, 192, fc4_b, nullptr, 0, Xcat, 192, 512, 96, 192, 0);

  // ================= t-branch =================
  k_t2<<<dim3((512 * 512) / 256), dim3(256), 0, stream>>>(t, t2);

  const int dils0[6] = {1, 2, 4, 8, 16, 32};
  const int dils1[4] = {1, 2, 4, 8};
  const int dils2[2] = {1, 2};
  const int* dilsArr[3] = {dils0, dils1, dils2};
  const int nconv[3] = {6, 4, 2};
  const int soffs[3] = {0, 4, 8};

  int li = 20;
  for (int sc = 0; sc < 3; ++sc) {
    float* cur = E0;
    float* nxt = E1;
    for (int j = 0; j < nconv[sc]; ++j) {
      const float* w  = (const float*)d_in[li + 0];
      const float* b  = (const float*)d_in[li + 1];
      const float* g  = (const float*)d_in[li + 2];
      const float* be = (const float*)d_in[li + 3];
      li += 4;
      int d = dilsArr[sc][j];
      float* dst = (j == 0) ? cur : nxt;
      if (j == 0) {
        k_conv1<<<dim3((512 * 64 * 512) / 256), dim3(256), 0, stream>>>(t2, w, b, dst, d);
      } else {
        k_conv_wmma<<<dim3(32, 512), dim3(128), 0, stream>>>(cur, w, b, dst, d);
      }
      k_bnstats<<<dim3(64), dim3(256), 0, stream>>>(dst, bn_m, bn_v);
      k_bnapply<<<dim3((512 * 64 * 512) / 256), dim3(256), 0, stream>>>(dst, bn_m, bn_v, g, be);
      if (j != 0) { float* tmp = cur; cur = nxt; nxt = tmp; }
    }
    const float* pool_w = (const float*)d_in[li + 0];
    const float* pool_b = (const float*)d_in[li + 1];
    li += 2;
    k_pool<<<dim3(512), dim3(256), 0, stream>>>(cur, pool_w, pool_b, nodes, soffs[sc]);
  }

  k_dtw_fuse<<<dim3(512), dim3(64), 0, stream>>>(nodes, protos, mem_vals, fused);
  // rp1 + gelu
  gemm(stream, fused, 128, rp1_w, 128, rp1_b, nullptr, 0, t_mid, 64, 512, 64, 128, 1);
  // rp2 -> t_out into Xcat[:, 96:192]
  gemm(stream, t_mid, 64, rp2_w, 64, rp2_b, nullptr, 0, Xcat + 96, 192, 512, 96, 64, 0);
  // fc8
  gemm(stream, Xcat, 192, fc8_w, 192, fc8_b, nullptr, 0, out_mat, 96, 512, 96, 192, 0);
  k_final<<<dim3((16 * 96 * 32 + 255) / 256), dim3(256), 0, stream>>>(out_mat, (float*)d_out);
}
